// WindowAttention_18897856102705
// MI455X (gfx1250) — matrix operand order repaired and checked by reading, not compiled, not run on hardware
//
#include <hip/hip_runtime.h>

// CDNA5 / gfx1250 wave32 WMMA implementation of Swin window attention.
// One workgroup (512 threads = 16 waves) per window; fully fused:
//   pre-kernel: transpose+convert Wqkv/Wproj -> f16 [N][K] in d_ws (2 MB)
//   phase 0: stage x -> LDS f16 (+ rpi, bias_table)
//   phase 1: QKV GEMM (WMMA f16, f32 accum) -> q (scaled), k, v^T in LDS
//   phase 2: per-head attention (1 wave per head): scores WMMA + bias +
//            masked softmax (shfl reductions) + attn@V WMMA -> out in LDS
//   phase 3: output projection GEMM (+bias) -> global f32

typedef __attribute__((ext_vector_type(16))) _Float16 v16h;
typedef __attribute__((ext_vector_type(8)))  _Float16 v8h;
typedef __attribute__((ext_vector_type(8)))  float    v8f;

#define NWIN   49
#define NPAD   64
#define CDIM   512
#define NHEADS 16
#define HDIM   32
#define QKVC   1536
#define BIAS_ROWS 169
#define NBLOCKS 4096

__device__ __forceinline__ v8f wmma_f16(v16h a, v16h b, v8f c) {
  // D = A(16x32 f16) * B(32x16 f16) + C(16x16 f32)
  return __builtin_amdgcn_wmma_f32_16x16x32_f16(false, a, false, b, (short)0, c,
                                                false, false);
}

// A-operand (16x32 f16) from row-major LDS [rows][stride].
// Layout: lane L holds row M=L%16; elems e -> K = k0+16*hf+e (hf = L/16),
// matching the B-operand k order.
__device__ __forceinline__ v16h load_a_lds(const _Float16* p, int stride,
                                           int row0, int k0, int l16, int hf) {
  const _Float16* r = p + (row0 + l16) * stride + k0 + 16 * hf;
  v8h lo = *(const v8h*)(r);
  v8h hi = *(const v8h*)(r + 8);
  v16h out;
#pragma unroll
  for (int i = 0; i < 8; ++i) { out[i] = lo[i]; out[i + 8] = hi[i]; }
  return out;
}

// B-operand (32x16 f16): lane L holds column N=col0+L%16, elems e -> K = k0+16*hf+e.
// p laid out with K contiguous per column: p[(col)*stride + K].
__device__ __forceinline__ v16h load_b_lds(const _Float16* p, int stride,
                                           int col0, int k0, int l16, int hf) {
  return *(const v16h*)(p + (col0 + l16) * stride + k0 + 16 * hf);
}

// B-operand from pre-transposed f16 global weight WT[N][CDIM]:
// one contiguous 32B load per lane (global_load_b128 pair).
__device__ __forceinline__ v16h load_b_gth(const _Float16* wt, int col, int k0,
                                           int hf) {
  return *(const v16h*)(wt + (size_t)col * CDIM + k0 + 16 * hf);
}

// Fallback: B-operand gathered from row-major f32 global weight W[K][N].
__device__ __forceinline__ v16h load_b_gf32(const float* p, int ldn, int col,
                                            int k0, int hf) {
  const float* q = p + (size_t)(k0 + 16 * hf) * ldn + col;
  v16h out;
#pragma unroll
  for (int e = 0; e < 16; ++e) out[e] = (_Float16)q[(size_t)e * ldn];
  return out;
}

// Transpose + f32->f16 convert both weights into workspace.
__global__ __launch_bounds__(256)
void convert_weights(const float* __restrict__ Wqkv, const float* __restrict__ Wproj,
                     _Float16* __restrict__ WqkvT, _Float16* __restrict__ WprojT) {
  int i = blockIdx.x * 256 + threadIdx.x;
  if (i < QKVC * CDIM) {                 // Wqkv[k][n] -> WqkvT[n][k]
    int k = i / QKVC, n = i % QKVC;      // coalesced read over n
    WqkvT[(size_t)n * CDIM + k] = (_Float16)Wqkv[i];
  } else {
    int j = i - QKVC * CDIM;
    if (j < CDIM * CDIM) {               // Wproj[k][n] -> WprojT[n][k]
      int k = j / CDIM, n = j % CDIM;
      WprojT[(size_t)n * CDIM + k] = (_Float16)Wproj[j];
    }
  }
}

template <bool TW>
__global__ __launch_bounds__(512)
void win_attn_gfx1250(const float* __restrict__ x, const float* __restrict__ Wqkv,
                      const float* __restrict__ Wproj, const float* __restrict__ bproj,
                      const float* __restrict__ bias_table, const int* __restrict__ rpi,
                      const _Float16* __restrict__ WqkvT,
                      const _Float16* __restrict__ WprojT,
                      float* __restrict__ out) {
  // ---- LDS (~315 KB of the 320 KB WGP budget) ----
  __shared__ _Float16 s_x[NPAD * CDIM];           // 64 KB; aliased as attn-out later
  __shared__ _Float16 s_q[NPAD * CDIM];           // 64 KB (pre-scaled by d^-1/2)
  __shared__ _Float16 s_k[NPAD * CDIM];           // 64 KB
  __shared__ _Float16 s_vt[CDIM * NPAD];          // 64 KB, vt[c][m] = v[m][c]
  __shared__ _Float16 s_attn[NHEADS * 16 * NPAD]; // 32 KB per-wave staging
  __shared__ int      s_rpi[NWIN * NWIN];         // 9.6 KB
  __shared__ float    s_bias[BIAS_ROWS * NHEADS]; // 10.8 KB

  const int tid  = threadIdx.x;
  const int lane = tid & 31;
  const int wave = __builtin_amdgcn_readfirstlane(tid >> 5);  // wave-uniform SGPR
  const int l16  = lane & 15;
  const int hf   = lane >> 4;
  const int win  = blockIdx.x;

  // ---------------- Phase 0: zero pads, stage inputs ----------------
  {
    v8h z = {};
    _Float16* regions[4] = {s_x, s_q, s_k, s_vt};
#pragma unroll
    for (int j = 0; j < 4; ++j) {
      v8h* r = (v8h*)regions[j];
      for (int i = tid; i < (NPAD * CDIM) / 8; i += 512) r[i] = z;
    }
  }
  __syncthreads();  // zeros visible before f16 staging overwrites rows < 49

  {
    const float* xw = x + (size_t)win * NWIN * CDIM;
    for (int i = tid; i < NWIN * CDIM; i += 512) s_x[i] = (_Float16)xw[i];
    for (int i = tid; i < NWIN * NWIN; i += 512) s_rpi[i] = rpi[i];
    for (int i = tid; i < BIAS_ROWS * NHEADS; i += 512) s_bias[i] = bias_table[i];
  }
  __syncthreads();

  // ---------------- Phase 1: QKV GEMM (x[64,512] @ Wqkv[512,1536]) ----------------
  // 96 N-tiles of 16 cols; each wave owns 6. K loop: 16 steps of 32.
  const float qscale = 0.17677669529663687f;  // 1/sqrt(32)
  for (int nti = 0; nti < 6; ++nti) {
    const int nt  = __builtin_amdgcn_readfirstlane(wave * 6 + nti);
    const int gc0 = nt * 16;
    v8f acc[4] = {};
    for (int kk = 0; kk < 16; ++kk) {
      v16h b;
      if constexpr (TW) b = load_b_gth(WqkvT, gc0 + l16, kk * 32, hf);
      else              b = load_b_gf32(Wqkv, QKVC, gc0 + l16, kk * 32, hf);
#pragma unroll
      for (int mt = 0; mt < 4; ++mt) {
        v16h a = load_a_lds(s_x, CDIM, mt * 16, kk * 32, l16, hf);
        acc[mt] = wmma_f16(a, b, acc[mt]);
      }
    }
    // D layout: reg r, lane -> row = mt*16 + r + 8*hf, col = gc0 + l16
    if (nt < 32) {            // -> q, pre-scaled
#pragma unroll
      for (int mt = 0; mt < 4; ++mt)
#pragma unroll
        for (int r = 0; r < 8; ++r) {
          int row = mt * 16 + r + 8 * hf;
          s_q[row * CDIM + gc0 + l16] = (_Float16)(acc[mt][r] * qscale);
        }
    } else if (nt < 64) {     // -> k
#pragma unroll
      for (int mt = 0; mt < 4; ++mt)
#pragma unroll
        for (int r = 0; r < 8; ++r) {
          int row = mt * 16 + r + 8 * hf;
          s_k[row * CDIM + (gc0 - CDIM) + l16] = (_Float16)acc[mt][r];
        }
    } else {                  // -> v transposed: vt[c][m]
      const int c = gc0 - 2 * CDIM + l16;
#pragma unroll
      for (int mt = 0; mt < 4; ++mt)
#pragma unroll
        for (int r = 0; r < 8; ++r) {
          int row = mt * 16 + r + 8 * hf;
          s_vt[c * NPAD + row] = (_Float16)acc[mt][r];
        }
    }
  }
  __syncthreads();

  // ---------------- Phase 2: attention, one wave per head ----------------
  {
    const int h = wave;
    const _Float16* qh  = s_q + h * HDIM;
    const _Float16* kh  = s_k + h * HDIM;
    const _Float16* vth = s_vt + h * HDIM * NPAD;
    _Float16* asw = s_attn + wave * 16 * NPAD;  // per-wave [16][64] staging
    _Float16* s_o = s_x;                        // alias: x no longer needed

    for (int mt = 0; mt < 4; ++mt) {
      // scores tile row-block: S = q_h[mt*16..+16, :32] @ k_h^T  (single K step)
      v8f sc[4];
      {
        v16h a = load_a_lds(qh, CDIM, mt * 16, 0, l16, hf);
#pragma unroll
        for (int nt = 0; nt < 4; ++nt) {
          v16h b = load_b_lds(kh, CDIM, nt * 16, 0, l16, hf);
          v8f z = {};
          sc[nt] = wmma_f16(a, b, z);
        }
      }
      // bias + mask + row softmax (rows live in reg r / half hf; cols across
      // 16 lanes x 4 nt registers)
#pragma unroll
      for (int r = 0; r < 8; ++r) {
        const int n = mt * 16 + r + 8 * hf;   // query row
        float mx = -3.0e38f;
#pragma unroll
        for (int nt = 0; nt < 4; ++nt) {
          const int m = nt * 16 + l16;        // key col
          float v = sc[nt][r];
          if (n < NWIN && m < NWIN) v += s_bias[s_rpi[n * NWIN + m] * NHEADS + h];
          if (m >= NWIN) v = -1.0e30f;
          sc[nt][r] = v;
          mx = fmaxf(mx, v);
        }
#pragma unroll
        for (int d = 1; d < 16; d <<= 1) mx = fmaxf(mx, __shfl_xor(mx, d, 32));
        float sum = 0.0f;
#pragma unroll
        for (int nt = 0; nt < 4; ++nt) {
          float e = __expf(sc[nt][r] - mx);
          sc[nt][r] = e;
          sum += e;
        }
#pragma unroll
        for (int d = 1; d < 16; d <<= 1) sum += __shfl_xor(sum, d, 32);
        const float inv = 1.0f / sum;
#pragma unroll
        for (int nt = 0; nt < 4; ++nt) sc[nt][r] *= inv;
      }
      // stage attn (f16) so it can be re-read in A-operand layout
#pragma unroll
      for (int nt = 0; nt < 4; ++nt)
#pragma unroll
        for (int r = 0; r < 8; ++r) {
          int rl = r + 8 * hf;
          asw[rl * NPAD + nt * 16 + l16] = (_Float16)sc[nt][r];
        }
      // out_tile = attn[16,64] @ v_h[64,32]   (2 K steps x 2 N tiles)
      v8f o[2] = {};
#pragma unroll
      for (int kk = 0; kk < 2; ++kk) {
        v16h a = load_a_lds(asw, NPAD, 0, kk * 32, l16, hf);
#pragma unroll
        for (int dt = 0; dt < 2; ++dt) {
          v16h b = load_b_lds(vth, NPAD, dt * 16, kk * 32, l16, hf);
          o[dt] = wmma_f16(a, b, o[dt]);
        }
      }
#pragma unroll
      for (int dt = 0; dt < 2; ++dt)
#pragma unroll
        for (int r = 0; r < 8; ++r) {
          int row = mt * 16 + r + 8 * hf;
          s_o[row * CDIM + h * HDIM + dt * 16 + l16] = (_Float16)o[dt][r];
        }
    }
  }
  __syncthreads();

  // ---------------- Phase 3: projection (out[64,512] @ Wproj[512,512] + b) ----------------
  // 32 N-tiles; each wave owns 2.
  {
    const _Float16* s_o = s_x;
    for (int nti = 0; nti < 2; ++nti) {
      const int nt  = __builtin_amdgcn_readfirstlane(wave * 2 + nti);
      const int gc0 = nt * 16;
      const float bv = bproj[gc0 + l16];
      v8f acc[4];
#pragma unroll
      for (int mt = 0; mt < 4; ++mt)
#pragma unroll
        for (int r = 0; r < 8; ++r) acc[mt][r] = bv;
      for (int kk = 0; kk < 16; ++kk) {
        v16h b;
        if constexpr (TW) b = load_b_gth(WprojT, gc0 + l16, kk * 32, hf);
        else              b = load_b_gf32(Wproj, CDIM, gc0 + l16, kk * 32, hf);
#pragma unroll
        for (int mt = 0; mt < 4; ++mt) {
          v16h a = load_a_lds(s_o, CDIM, mt * 16, kk * 32, l16, hf);
          acc[mt] = wmma_f16(a, b, acc[mt]);
        }
      }
#pragma unroll
      for (int mt = 0; mt < 4; ++mt)
#pragma unroll
        for (int r = 0; r < 8; ++r) {
          int row = mt * 16 + r + 8 * hf;
          if (row < NWIN)
            out[((size_t)win * NWIN + row) * CDIM + gc0 + l16] = acc[mt][r];
        }
    }
  }
}

extern "C" void kernel_launch(void* const* d_in, const int* in_sizes, int n_in,
                              void* d_out, int out_size, void* d_ws, size_t ws_size,
                              hipStream_t stream) {
  (void)in_sizes; (void)n_in; (void)out_size;
  const float* x          = (const float*)d_in[0];
  const float* Wqkv       = (const float*)d_in[1];
  const float* Wproj      = (const float*)d_in[2];
  const float* bproj      = (const float*)d_in[3];
  const float* bias_table = (const float*)d_in[4];
  const int*   rpi        = (const int*)d_in[5];
  float* outp             = (float*)d_out;

  const size_t nWqkvT = (size_t)QKVC * CDIM;          // 786432 halves
  const size_t nWprojT = (size_t)CDIM * CDIM;         // 262144 halves
  const size_t needBytes = (nWqkvT + nWprojT) * 2;    // 2 MB

  if (d_ws != nullptr && ws_size >= needBytes) {
    _Float16* WqkvT  = (_Float16*)d_ws;
    _Float16* WprojT = (_Float16*)((char*)d_ws + nWqkvT * 2);
    const int total = (int)(nWqkvT + nWprojT);
    convert_weights<<<(total + 255) / 256, 256, 0, stream>>>(Wqkv, Wproj,
                                                             WqkvT, WprojT);
    win_attn_gfx1250<true><<<dim3(NBLOCKS), dim3(512), 0, stream>>>(
        x, Wqkv, Wproj, bproj, bias_table, rpi, WqkvT, WprojT, outp);
  } else {
    win_attn_gfx1250<false><<<dim3(NBLOCKS), dim3(512), 0, stream>>>(
        x, Wqkv, Wproj, bproj, bias_table, rpi, nullptr, nullptr, outp);
  }
}
